// SelfAttention_9560597201229
// MI455X (gfx1250) — matrix operand order repaired and checked by reading, not compiled, not run on hardware
//
#include <hip/hip_runtime.h>

// ---------- POD vector types ----------
typedef __attribute__((ext_vector_type(4)))  unsigned int v4u;
typedef __attribute__((ext_vector_type(2)))  unsigned int v2u;
typedef __attribute__((ext_vector_type(4)))  int          v4i;
typedef __attribute__((ext_vector_type(4)))  float        v4f;
typedef __attribute__((ext_vector_type(16))) __bf16       v16bf;
typedef __attribute__((ext_vector_type(8)))  float        v8f;

union Frag {            // 32 bytes: one WMMA A/B operand (16 bf16 = 8 VGPRs)
    v4u   q[2];
    v16bf v;
};

// ---- optional CDNA5 async global->LDS path (ASYNCcnt), guarded ----
#if defined(__HIP_DEVICE_COMPILE__) && \
    __has_builtin(__builtin_amdgcn_global_load_async_to_lds_b128) && \
    __has_builtin(__builtin_amdgcn_s_wait_asynccnt)
#define USE_ASYNC_LDS 1
#else
#define USE_ASYNC_LDS 0
#endif

#if USE_ASYNC_LDS
typedef __attribute__((address_space(1))) v4i glb_v4i;   // clang prints AS1 as __device__
typedef __attribute__((address_space(3))) v4i lds_v4i;
__device__ __forceinline__ glb_v4i* to_gbl128(const void* p) {
    return (glb_v4i*)(__UINTPTR_TYPE__)p;                 // value-preserving 64-bit
}
__device__ __forceinline__ lds_v4i* to_lds128(void* p) {
    // generic LDS pointers carry the LDS byte offset in the low 32 bits
    return (lds_v4i*)(unsigned int)(__UINTPTR_TYPE__)p;
}
#endif

__device__ __forceinline__ unsigned short f32_to_bf16(float f) {
    unsigned int u = __float_as_uint(f);
    u += 0x7FFFu + ((u >> 16) & 1u);   // round-to-nearest-even
    return (unsigned short)(u >> 16);
}

// ---------------- fp32 -> bf16 convert (bandwidth-bound pre-pass) -------------
__global__ __launch_bounds__(256) void cvt_f32_bf16(const float* __restrict__ in,
                                                    unsigned short* __restrict__ out,
                                                    int n4) {
    int i = (blockIdx.x * 256 + threadIdx.x);
    if (i >= n4) return;
    v4f v = *(const v4f*)(in + (size_t)i * 4);
    v2u p;
    p.x = (unsigned)f32_to_bf16(v.x) | ((unsigned)f32_to_bf16(v.y) << 16);
    p.y = (unsigned)f32_to_bf16(v.z) | ((unsigned)f32_to_bf16(v.w) << 16);
    *(v2u*)(out + (size_t)i * 4) = p;
}

// ------------- 64x64 LDS-tiled transpose, SrcT in {float, ushort} -> bf16 -----
// out[c][r] = bf16(in[r][c]).  grid: x = C/64, y = R/64, z = batch.
template <typename SrcT>
__global__ __launch_bounds__(256)
void transpose_to_bf16(const SrcT* __restrict__ in, long long strideIn, int ldin,
                       unsigned short* __restrict__ out, long long strideOut, int ldout) {
    __shared__ unsigned short tile[64][72];     // +8 halfs padding
    const int tid = threadIdx.x;
    const int rbase = blockIdx.y * 64, cbase = blockIdx.x * 64;

    {
        int r = tid >> 2, cseg = (tid & 3) * 16;
        const SrcT* src = in + blockIdx.z * strideIn + (size_t)(rbase + r) * ldin + cbase + cseg;
        unsigned short h[16];
        if constexpr (sizeof(SrcT) == 4) {
            v4f f0 = *(const v4f*)(src + 0), f1 = *(const v4f*)(src + 4);
            v4f f2 = *(const v4f*)(src + 8), f3 = *(const v4f*)(src + 12);
#pragma unroll
            for (int i = 0; i < 4; ++i) {
                h[i]      = f32_to_bf16(f0[i]);
                h[4 + i]  = f32_to_bf16(f1[i]);
                h[8 + i]  = f32_to_bf16(f2[i]);
                h[12 + i] = f32_to_bf16(f3[i]);
            }
        } else {
            *(v4u*)&h[0] = *(const v4u*)src;
            *(v4u*)&h[8] = *(const v4u*)(src + 8);
        }
        v4u* d = (v4u*)&tile[r][cseg];
        d[0] = *(v4u*)&h[0];
        d[1] = *(v4u*)&h[8];
    }
    __syncthreads();

    {
        int c = tid >> 2, rseg = (tid & 3) * 16;
        unsigned short h[16];
#pragma unroll
        for (int i = 0; i < 16; ++i) h[i] = tile[rseg + i][c];
        unsigned short* dst = out + blockIdx.z * strideOut + (size_t)(cbase + c) * ldout + rbase + rseg;
        *(v4u*)dst       = *(v4u*)&h[0];
        *(v4u*)(dst + 8) = *(v4u*)&h[8];
    }
}

// ---------------- bf16 WMMA GEMM ---------------------------------------------
// C[M,N] = scale * A[M,K] x B^T, with B stored [N,K] row-major (ldb).
// Block tile 128x256, BK=32. 8 waves in 2x4 grid; each wave owns 64x64 =
// 4x4 WMMA tiles -> 16 v_wmma per K-stage.  Double-buffered LDS; staging is
// either GLOBAL_LOAD_ASYNC_TO_LDS_B128 (ASYNCcnt path, if toolchain exposes
// the builtin) or global_load_b128 -> register -> ds_store_b128 fallback.
constexpr int LDSS = 40;                 // 32 + 8 halfs padding (bank conflicts)
constexpr int BM = 128, BN = 256, BK = 32;

__device__ __forceinline__ void ldg_a(const unsigned short* Ab, int lda, int k0,
                                      int tid, v4u r[2]) {
    int row = tid >> 1, seg = (tid & 1) * 16;
    const v4u* s = (const v4u*)(Ab + (size_t)row * lda + k0 + seg);
    r[0] = s[0]; r[1] = s[1];
}
__device__ __forceinline__ void sts_a(unsigned short* As, int tid, const v4u r[2]) {
    int row = tid >> 1, seg = (tid & 1) * 16;
    v4u* d = (v4u*)&As[row * LDSS + seg];
    d[0] = r[0]; d[1] = r[1];
}
__device__ __forceinline__ void ldg_b(const unsigned short* Bb, int ldb, int k0,
                                      int tid, v4u r[4]) {
    const v4u* s = (const v4u*)(Bb + (size_t)tid * ldb + k0);    // row n = tid
    r[0] = s[0]; r[1] = s[1]; r[2] = s[2]; r[3] = s[3];
}
__device__ __forceinline__ void sts_b(unsigned short* Bs, int tid, const v4u r[4]) {
    v4u* d = (v4u*)&Bs[tid * LDSS];
    d[0] = r[0]; d[1] = r[1]; d[2] = r[2]; d[3] = r[3];
}

template <bool OUT_BF16>
__global__ __launch_bounds__(256)
void gemm_bf16_wmma(const unsigned short* __restrict__ A, long long strideA, int lda,
                    const unsigned short* __restrict__ B, long long strideB, int ldb,
                    void* __restrict__ C, long long strideC, int ldc,
                    int K, float scale) {
    __shared__ unsigned short As[2][BM * LDSS];   // 2 x 5120 halfs
    __shared__ unsigned short Bs[2][BN * LDSS];   // 2 x 10240 halfs  (60 KB total)

    const int tid  = threadIdx.x;
    const int lane = tid & 31;
    const int wave = tid >> 5;
    const int wr   = wave >> 2;            // 0..1  (64-row band)
    const int wc   = wave & 3;             // 0..3  (64-col band)

    const unsigned short* Ab = A + (size_t)blockIdx.z * strideA
                                 + (size_t)(blockIdx.y * BM) * lda;
    const unsigned short* Bb = B + (size_t)blockIdx.z * strideB
                                 + (size_t)(blockIdx.x * BN) * ldb;

    v8f acc[4][4] = {};

#if USE_ASYNC_LDS
    // ---- async DMA staging: memory -> LDS directly, tracked by ASYNCcnt ----
    const int arow_st = tid >> 1, aseg_st = (tid & 1) * 16;
    const unsigned short* agp = Ab + (size_t)arow_st * lda + aseg_st;
    const unsigned short* bgp = Bb + (size_t)tid * ldb;
    unsigned short* alp0 = &As[0][arow_st * LDSS + aseg_st];
    unsigned short* blp0 = &Bs[0][tid * LDSS];
    unsigned short* alp1 = &As[1][arow_st * LDSS + aseg_st];
    unsigned short* blp1 = &Bs[1][tid * LDSS];

    auto issue_stage = [&](int k0, int b) {
        unsigned short* al = b ? alp1 : alp0;
        unsigned short* bl = b ? blp1 : blp0;
        __builtin_amdgcn_global_load_async_to_lds_b128(to_gbl128(agp + k0),
                                                       to_lds128(al), 0, 0);
        __builtin_amdgcn_global_load_async_to_lds_b128(to_gbl128(agp + k0 + 8),
                                                       to_lds128(al + 8), 0, 0);
#pragma unroll
        for (int j = 0; j < 4; ++j)
            __builtin_amdgcn_global_load_async_to_lds_b128(to_gbl128(bgp + k0 + j * 8),
                                                           to_lds128(bl + j * 8), 0, 0);
    };

    issue_stage(0, 0);
    __builtin_amdgcn_s_wait_asynccnt(0);
    __syncthreads();
#else
    v4u ra[2], rb[4];
    ldg_a(Ab, lda, 0, tid, ra);
    ldg_b(Bb, ldb, 0, tid, rb);
    sts_a(As[0], tid, ra);
    sts_b(Bs[0], tid, rb);
    __syncthreads();
#endif

    int buf = 0;
    for (int k0 = 0; k0 < K; k0 += BK) {
        const bool has_next = (k0 + BK) < K;
        if (has_next) {
#if USE_ASYNC_LDS
            issue_stage(k0 + BK, buf ^ 1);
#else
            ldg_a(Ab, lda, k0 + BK, tid, ra);
            ldg_b(Bb, ldb, k0 + BK, tid, rb);
#endif
            if (k0 + 2 * BK < K) {   // warm L2/WGP$ two stages ahead
                __builtin_prefetch(Ab + (size_t)(tid >> 1) * lda + k0 + 2 * BK, 0, 0);
                __builtin_prefetch(Bb + (size_t)tid * ldb + k0 + 2 * BK, 0, 0);
            }
        }

        // ---- fragments from current buffer ----
        Frag a[4], b[4];
        const int arow = wr * 64 + (lane & 15);
        const int kb8  = (lane >> 4) * 8;       // A: lanes 16-31 K=8../24..
#pragma unroll
        for (int t = 0; t < 4; ++t) {
            const unsigned short* p = &As[buf][(arow + t * 16) * LDSS];
            a[t].q[0] = *(const v4u*)(p + kb8);
            a[t].q[1] = *(const v4u*)(p + 16 + kb8);
        }
        const int bcol = wc * 64 + (lane & 15);
        const int kb16 = (lane >> 4) * 8;       // B: lanes 16-31 K=8../24..
#pragma unroll
        for (int u = 0; u < 4; ++u) {
            const unsigned short* p = &Bs[buf][(bcol + u * 16) * LDSS + kb16];
            b[u].q[0] = *(const v4u*)p;
            b[u].q[1] = *(const v4u*)(p + 16);
        }

#pragma unroll
        for (int t = 0; t < 4; ++t)
#pragma unroll
            for (int u = 0; u < 4; ++u)
                acc[t][u] = __builtin_amdgcn_wmma_f32_16x16x32_bf16(
                    false, a[t].v, false, b[u].v, (short)0, acc[t][u],
                    false, false);

        if (has_next) {
#if USE_ASYNC_LDS
            __builtin_amdgcn_s_wait_asynccnt(0);
#else
            sts_a(As[buf ^ 1], tid, ra);
            sts_b(Bs[buf ^ 1], tid, rb);
#endif
            __syncthreads();     // readers of buf^1 wait here next iteration
        }
        buf ^= 1;
    }

    // ---- epilogue: 16x16 f32 C/D layout (lane = N, VGPR v = M row) ----
    const long long cbase = (long long)blockIdx.z * strideC;
    const int col  = lane & 15;
    const int rsel = (lane >> 4) * 8;
#pragma unroll
    for (int t = 0; t < 4; ++t) {
#pragma unroll
        for (int u = 0; u < 4; ++u) {
            int m0 = blockIdx.y * BM + wr * 64 + t * 16 + rsel;
            int n0 = blockIdx.x * BN + wc * 64 + u * 16 + col;
#pragma unroll
            for (int v = 0; v < 8; ++v) {
                float val = acc[t][u][v] * scale;
                long long idx = cbase + (long long)(m0 + v) * ldc + n0;
                if (OUT_BF16)
                    ((unsigned short*)C)[idx] = f32_to_bf16(val);
                else
                    ((float*)C)[idx] = val;
            }
        }
    }
}

// ---------------- row softmax: fp32 in, bf16 probabilities out ---------------
__global__ __launch_bounds__(256)
void softmax_rows_2048(const float* __restrict__ S, unsigned short* __restrict__ P) {
    __shared__ float red[256];
    const int row = blockIdx.x;
    const float* src = S + (size_t)row * 2048;
    unsigned short* dst = P + (size_t)row * 2048;

    float x[8];
#pragma unroll
    for (int i = 0; i < 8; ++i) x[i] = src[threadIdx.x + i * 256];

    float lmax = x[0];
#pragma unroll
    for (int i = 1; i < 8; ++i) lmax = fmaxf(lmax, x[i]);
    red[threadIdx.x] = lmax;
    __syncthreads();
    for (int s = 128; s > 0; s >>= 1) {
        if (threadIdx.x < s) red[threadIdx.x] = fmaxf(red[threadIdx.x], red[threadIdx.x + s]);
        __syncthreads();
    }
    const float rmax = red[0];
    __syncthreads();

    float e[8], lsum = 0.f;
#pragma unroll
    for (int i = 0; i < 8; ++i) { e[i] = __expf(x[i] - rmax); lsum += e[i]; }
    red[threadIdx.x] = lsum;
    __syncthreads();
    for (int s = 128; s > 0; s >>= 1) {
        if (threadIdx.x < s) red[threadIdx.x] += red[threadIdx.x + s];
        __syncthreads();
    }
    const float inv = 1.0f / red[0];
#pragma unroll
    for (int i = 0; i < 8; ++i)
        dst[threadIdx.x + i * 256] = f32_to_bf16(e[i] * inv);
}

// ------------------------------- launcher ------------------------------------
extern "C" void kernel_launch(void* const* d_in, const int* in_sizes, int n_in,
                              void* d_out, int out_size, void* d_ws, size_t ws_size,
                              hipStream_t stream) {
    constexpr int B = 4, S = 2048, E = 1024;
    constexpr long long MS = (long long)B * S;      // 8192 total rows

    const float* X = (const float*)d_in[0];         // [B,S,E] fp32
    const float* W = (const float*)d_in[1];         // [E,3E]  fp32
    float* out = (float*)d_out;                     // [B,S,E] fp32

    // workspace layout (all 16B aligned)
    unsigned short* Xb  = (unsigned short*)d_ws;                 // 8192x1024      16 MB
    unsigned short* WT  = Xb  + (size_t)MS * E;                  // 3072x1024       6 MB
    unsigned short* QKV = WT  + (size_t)3 * E * E;               // 8192x3072      48 MB
    unsigned short* VT  = QKV + (size_t)MS * 3 * E;              // 4x1024x2048    16 MB
    unsigned short* Pb  = VT  + (size_t)B * E * S;               // 4x2048x2048    32 MB
    float*          Sc  = (float*)(Pb + (size_t)B * S * S);      // 4x2048x2048    64 MB

    // 1) X -> bf16 ; W -> bf16 transposed [3E, E]
    {
        int n4 = (int)(MS * E / 4);
        cvt_f32_bf16<<<(n4 + 255) / 256, 256, 0, stream>>>(X, Xb, n4);
        transpose_to_bf16<float><<<dim3(3 * E / 64, E / 64, 1), 256, 0, stream>>>(
            W, 0, 3 * E, WT, 0, E);
    }

    // 2) QKV = Xb @ WT^T -> bf16 [8192, 3072]; Q/K/V are column views (ld 3072)
    gemm_bf16_wmma<true><<<dim3(3 * E / BN, MS / BM, 1), 256, 0, stream>>>(
        Xb, 0, E, WT, 0, E, QKV, 0, 3 * E, E, 1.0f);

    // 3) V[b] (2048x1024, ld 3072) -> VT[b] (1024x2048)
    transpose_to_bf16<unsigned short><<<dim3(E / 64, S / 64, B), 256, 0, stream>>>(
        QKV + 2 * E, (long long)S * 3 * E, 3 * E, VT, (long long)E * S, S);

    // 4) Sc[b] = (Q[b] @ K[b]^T) / 32 ; K rows are [n,e] row-major already
    gemm_bf16_wmma<false><<<dim3(S / BN, S / BM, B), 256, 0, stream>>>(
        QKV + 0,  (long long)S * 3 * E, 3 * E,
        QKV + E,  (long long)S * 3 * E, 3 * E,
        Sc,       (long long)S * S,     S,
        E, 0.03125f);

    // 5) row softmax -> bf16 P
    softmax_rows_2048<<<B * S, 256, 0, stream>>>(Sc, Pb);

    // 6) out[b] = P[b] @ VT[b]^T  (fp32 result)
    gemm_bf16_wmma<false><<<dim3(E / BN, S / BM, B), 256, 0, stream>>>(
        Pb, (long long)S * S, S,
        VT, (long long)E * S, S,
        out, (long long)S * E, E,
        S, 1.0f);
}
